// FactMachine_78391743087031
// MI455X (gfx1250) — matrix operand order repaired and checked by reading, not compiled, not run on hardware
//
#include <hip/hip_runtime.h>
#include <hip/hip_bf16.h>

// Factorization machine, MI455X (gfx1250, wave32, WMMA).
// Split-bf16 (hi+lo) GEMM X @ Vmat via v_wmma_f32_16x16x32_bf16
// (Xh*Vh + Xh*Vl + Xl*Vh -> ~fp32 accuracy at bf16 matrix rates).
// X (64 MB) streamed exactly once; bf16 split/swizzle happens in registers
// at staging time; scalar row-dots (X.w, X^2.sqnorm) fused into the stager.
// Global loads for chunk kc+1 are software-pipelined under the WMMA work of
// chunk kc; WMMA issue order is interleaved across the 4 independent
// accumulators. X is addressed directly off the kernel arg so the loads
// lower to global_load_b128 (LOADcnt only), not flat_load (LOADcnt+DScnt).

typedef __attribute__((ext_vector_type(16))) __bf16 v16bf;
typedef __attribute__((ext_vector_type(4)))  __bf16 v4bf;
typedef __attribute__((ext_vector_type(8)))  float  v8f;

#define FM_M 32768
#define FM_D 512
#define FM_F 64

#define WMMA_BF16(A, B, C) \
  __builtin_amdgcn_wmma_f32_16x16x32_bf16(false, (A), false, (B), (short)0, (C), false, false)

// ---------------------------------------------------------------------------
// Prep A: Vmat = V_w^T + V_b, split to bf16 hi/lo, stored in WMMA B-fragment
// order: index = (((kc*4 + tile)*32 + lane)*16 + j)
//   lane 0-15 : N = lane,       K = kc*32 + j        (j = 0..15)
//   lane 16-31: N = lane - 16,  K = kc*32 + 16 + j
// ---------------------------------------------------------------------------
__global__ __launch_bounds__(256) void fm_prep_vfrag(
    const float* __restrict__ V_w, const float* __restrict__ V_b,
    __bf16* __restrict__ Vh, __bf16* __restrict__ Vl)
{
  int idx = blockIdx.x * 256 + threadIdx.x;   // 0 .. 32767
  int j  = idx & 15;
  int l  = (idx >> 4) & 31;
  int t  = (idx >> 9) & 3;
  int kc = idx >> 11;
  int K  = kc * 32 + (l >> 4) * 16 + j;
  int f  = t * 16 + (l & 15);
  float v = V_w[f * FM_D + K] + V_b[f];
  __bf16 h = (__bf16)v;
  Vh[idx] = h;
  Vl[idx] = (__bf16)(v - (float)h);
}

// ---------------------------------------------------------------------------
// Prep B: sq_norms[d] = sum_f Vmat[d,f]^2
// ---------------------------------------------------------------------------
__global__ __launch_bounds__(64) void fm_prep_sqn(
    const float* __restrict__ V_w, const float* __restrict__ V_b,
    float* __restrict__ sqn)
{
  __shared__ float red[64];
  int d = blockIdx.x;
  int f = threadIdx.x;
  float v = V_w[f * FM_D + d] + V_b[f];
  red[f] = v * v;
  __syncthreads();
  for (int s = 32; s > 0; s >>= 1) {
    if (f < s) red[f] += red[f + s];
    __syncthreads();
  }
  if (f == 0) sqn[d] = red[0];
}

// ---------------------------------------------------------------------------
// Main kernel: 256 threads (8 waves), 128 rows/block, full F=64 per wave.
// Dynamic LDS layout (bytes):
//   0      : sVh  B-frags hi [32768 bf16]  65536
//   65536  : sVl  B-frags lo [32768 bf16]  65536
//   131072 : sAh  A-frags hi [8 wv][32 ln][16] bf16   8192
//   139264 : sAl  A-frags lo                          8192
//   147456 : sWw  [512] f32   2048
//   149504 : sSq  [512] f32   2048
//   151552 : sLin [128][8] f32 4096
//   155648 : sSqd [128][8] f32 4096
//   159744 : sRow [128] f32     512
// total 160256 B  (2 workgroups fit in a 320KB WGP)
// ---------------------------------------------------------------------------
#define FM_SMEM 160256

__global__ __launch_bounds__(256) void fm_main(
    const float* __restrict__ X,
    const float* __restrict__ w0_w, const float* __restrict__ w0_b,
    const float* __restrict__ w_w,  const float* __restrict__ w_b,
    const __bf16* __restrict__ Vh,  const __bf16* __restrict__ Vl,
    const float* __restrict__ sqn,  float* __restrict__ out)
{
  extern __shared__ char smem[];
  __bf16* sVh = (__bf16*)(smem);
  __bf16* sVl = (__bf16*)(smem + 65536);
  __bf16* sAh = (__bf16*)(smem + 131072);
  __bf16* sAl = (__bf16*)(smem + 139264);
  float*  sWw = (float*) (smem + 147456);
  float*  sSq = (float*) (smem + 149504);
  float*  sLin= (float*) (smem + 151552);
  float*  sSqd= (float*) (smem + 155648);
  float*  sRow= (float*) (smem + 159744);

  const int tid  = threadIdx.x;
  const int lane = tid & 31;
  const int wave = tid >> 5;
  const int row0 = blockIdx.x * 128;

  // ---- Phase 0: V fragments + small vectors into LDS ----
  {
    const uint4* gVh = (const uint4*)Vh;
    const uint4* gVl = (const uint4*)Vl;
    uint4* lVh = (uint4*)sVh;
    uint4* lVl = (uint4*)sVl;
    #pragma unroll
    for (int i = 0; i < 16; ++i) {
      int q = i * 256 + tid;
      lVh[q] = gVh[q];
      lVl[q] = gVl[q];
    }
    #pragma unroll
    for (int i = 0; i < 2; ++i) {
      int q = i * 256 + tid;
      sWw[q] = w_w[q];
      sSq[q] = sqn[q];
    }
  }

  // ---- stager geometry (fixed per thread) ----
  // thread owns rows rbase+32*i (i=0..3), K columns cbase..cbase+3 per chunk
  const int rbase = tid >> 3;          // 0..31
  const int cbase = (tid & 7) << 2;    // 0,4,..,28
  // A-fragment target for (row r, K cbase..cbase+3):
  //   lane = (r&15) + 16*(cbase>>4),  j0 = cbase&15
  const int fl_lane = cbase >> 4;
  const int fj0     = cbase & 15;

  float linp[4] = {0.f, 0.f, 0.f, 0.f};
  float sqp [4] = {0.f, 0.f, 0.f, 0.f};
  v8f acc0 = {}, acc1 = {}, acc2 = {}, acc3 = {};

  const int arow  = wave * 32 + lane;      // A-fragment slot of this lane
  const v16bf* fVh = (const v16bf*)sVh;
  const v16bf* fVl = (const v16bf*)sVl;
  const v16bf* fAh = (const v16bf*)sAh;
  const v16bf* fAl = (const v16bf*)sAl;

  // fixed per-thread row offsets into X; koff advances 32 floats per chunk
  size_t rowoff[4];
  #pragma unroll
  for (int i = 0; i < 4; ++i)
    rowoff[i] = (size_t)(row0 + rbase + 32 * i) * FM_D + cbase;

  // ---- preload chunk 0 ----
  float4 xv[4];
  #pragma unroll
  for (int i = 0; i < 4; ++i)
    xv[i] = *(const float4*)(X + rowoff[i]);

  __syncthreads();   // covers phase-0 LDS fills

  for (int kc = 0; kc < 16; ++kc) {
    // ---- convert/store current chunk's X + fused scalar dots ----
    {
      const int kb = kc * 32 + cbase;
      float4 w4 = *(const float4*)(sWw + kb);   // broadcast reads
      float4 s4 = *(const float4*)(sSq + kb);
      #pragma unroll
      for (int i = 0; i < 4; ++i) {
        float4 x = xv[i];
        linp[i] = fmaf(x.x, w4.x, fmaf(x.y, w4.y, fmaf(x.z, w4.z, fmaf(x.w, w4.w, linp[i]))));
        sqp[i]  = fmaf(x.x * x.x, s4.x, fmaf(x.y * x.y, s4.y,
                  fmaf(x.z * x.z, s4.z, fmaf(x.w * x.w, s4.w, sqp[i]))));
        v4bf hi, lo;
        __bf16 h0 = (__bf16)x.x, h1 = (__bf16)x.y, h2 = (__bf16)x.z, h3 = (__bf16)x.w;
        hi[0] = h0; hi[1] = h1; hi[2] = h2; hi[3] = h3;
        lo[0] = (__bf16)(x.x - (float)h0);
        lo[1] = (__bf16)(x.y - (float)h1);
        lo[2] = (__bf16)(x.z - (float)h2);
        lo[3] = (__bf16)(x.w - (float)h3);
        int r  = rbase + 32 * i;
        int fo = (((r >> 4) * 32) + (r & 15) + 16 * fl_lane) * 16 + fj0;
        *(v4bf*)(sAh + fo) = hi;
        *(v4bf*)(sAl + fo) = lo;
      }
    }
    __syncthreads();   // fragments visible

    // ---- software-pipelined global loads for next chunk ----
    if (kc + 1 < 16) {
      const int koff = (kc + 1) * 32;
      #pragma unroll
      for (int i = 0; i < 4; ++i)
        xv[i] = *(const float4*)(X + rowoff[i] + koff);
      __builtin_prefetch(X + rowoff[0] + koff + 32, 0, 1);  // one chunk further
    }

    // ---- A fragments: pure b128 LDS loads ----
    v16bf a_hi = fAh[arow];
    v16bf a_lo = fAl[arow];

    // ---- B fragments (all 8 live) + 12 WMMAs interleaved across accs ----
    const int fb = kc * 4 * 32 + lane;
    v16bf bh0 = fVh[fb +  0], bh1 = fVh[fb + 32], bh2 = fVh[fb + 64], bh3 = fVh[fb + 96];
    v16bf bl0 = fVl[fb +  0], bl1 = fVl[fb + 32], bl2 = fVl[fb + 64], bl3 = fVl[fb + 96];

    acc0 = WMMA_BF16(a_hi, bh0, acc0);
    acc1 = WMMA_BF16(a_hi, bh1, acc1);
    acc2 = WMMA_BF16(a_hi, bh2, acc2);
    acc3 = WMMA_BF16(a_hi, bh3, acc3);

    acc0 = WMMA_BF16(a_hi, bl0, acc0);
    acc1 = WMMA_BF16(a_hi, bl1, acc1);
    acc2 = WMMA_BF16(a_hi, bl2, acc2);
    acc3 = WMMA_BF16(a_hi, bl3, acc3);

    acc0 = WMMA_BF16(a_lo, bh0, acc0);
    acc1 = WMMA_BF16(a_lo, bh1, acc1);
    acc2 = WMMA_BF16(a_lo, bh2, acc2);
    acc3 = WMMA_BF16(a_lo, bh3, acc3);

    __syncthreads();   // before next chunk overwrites A fragments
  }

  // ---- epilogue ----
  {
    const int c = tid & 7;
    #pragma unroll
    for (int i = 0; i < 4; ++i) {
      int r = rbase + 32 * i;
      sLin[r * 8 + c] = linp[i];
      sSqd[r * 8 + c] = sqp[i];
    }
  }

  // per-row sum of XV^2: C layout -> lane holds column (lane&15),
  // VGPR r holds row (lane>>4)*8 + r of this wave's 16-row tile.
  float s[8];
  #pragma unroll
  for (int r = 0; r < 8; ++r) {
    float a = acc0[r], b = acc1[r], c = acc2[r], d = acc3[r];
    s[r] = a * a + b * b + c * c + d * d;
  }
  #pragma unroll
  for (int off = 1; off < 16; off <<= 1) {
    #pragma unroll
    for (int r = 0; r < 8; ++r) s[r] += __shfl_xor(s[r], off, 32);
  }
  if ((lane & 15) == 0) {
    int base = wave * 16 + (lane >> 4) * 8;
    #pragma unroll
    for (int r = 0; r < 8; ++r) sRow[base + r] = s[r];
  }
  __syncthreads();

  if (tid < 128) {
    float c0 = w0_w[0] + w0_b[0] + w_b[0];
    float lin = 0.f, sqd = 0.f;
    #pragma unroll
    for (int c = 0; c < 8; ++c) {
      lin += sLin[tid * 8 + c];
      sqd += sSqd[tid * 8 + c];
    }
    out[row0 + tid] = c0 + lin + 0.5f * (sRow[tid] - sqd);
  }
}

// ---------------------------------------------------------------------------
extern "C" void kernel_launch(void* const* d_in, const int* in_sizes, int n_in,
                              void* d_out, int out_size, void* d_ws, size_t ws_size,
                              hipStream_t stream)
{
  const float* X    = (const float*)d_in[0];
  const float* w0_w = (const float*)d_in[1];
  const float* w0_b = (const float*)d_in[2];
  const float* w_w  = (const float*)d_in[3];
  const float* w_b  = (const float*)d_in[4];
  const float* V_w  = (const float*)d_in[5];
  const float* V_b  = (const float*)d_in[6];
  float* out = (float*)d_out;

  // workspace: Vh frag (64KB) | Vl frag (64KB) | sq_norms (2KB)
  __bf16* Vhp = (__bf16*)d_ws;
  __bf16* Vlp = (__bf16*)((char*)d_ws + 65536);
  float*  sqn = (float*)((char*)d_ws + 131072);

  fm_prep_vfrag<<<FM_D * FM_F / 256, 256, 0, stream>>>(V_w, V_b, Vhp, Vlp);
  fm_prep_sqn<<<FM_D, 64, 0, stream>>>(V_w, V_b, sqn);
  fm_main<<<FM_M / 128, 256, FM_SMEM, stream>>>(X, w0_w, w0_b, w_w, w_b,
                                                Vhp, Vlp, sqn, out);
}
